// MHSA_8323646620297
// MI455X (gfx1250) — compile-verified, matrix operand order checked by reading, not run on hardware
//
#include <hip/hip_runtime.h>

typedef __attribute__((ext_vector_type(16))) _Float16 v16h;
typedef __attribute__((ext_vector_type(8)))  _Float16 v8h;
typedef __attribute__((ext_vector_type(8)))  float    v8f;

#define BB    16
#define CC    512
#define WW    32
#define HH    32
#define NHEAD 8
#define DHD   64
#define NN    1024

// ---------------------------------------------------------------------------
// WMMA 16x16x32 f16 fragment loader (wave32, per CDNA5 ISA).
// Works for BOTH operands:
//   A tile stored [M][K] row-major  -> lane&15 = M row
//   B tile stored [N][K] (column-of-B per LDS row) -> lane&15 = N column
// Per-lane: halves 0..7 -> K=k0..k0+7, halves 8..15 -> K=k0+16..k0+23,
// k0 = (lane>=16) ? 8 : 0.  `ld` (halves) must be a multiple of 8, base 16B.
// ---------------------------------------------------------------------------
__device__ __forceinline__ v16h ld_frag(const _Float16* base, int ld) {
  const int lane = threadIdx.x & 31;
  const _Float16* p = base + (lane & 15) * ld + ((lane >> 4) << 3);
  v8h lo = *(const v8h*)(p);        // K = k0 .. k0+7
  v8h hi = *(const v8h*)(p + 16);   // K = k0+16 .. k0+23
  v16h a;
#pragma unroll
  for (int i = 0; i < 8; ++i) { a[i] = lo[i]; a[8 + i] = hi[i]; }
  return a;
}

// ---------------------------------------------------------------------------
// Async global->LDS 16B copy (CDNA5 GLOBAL_LOAD_ASYNC_TO_LDS_B128, ASYNCcnt).
// Per ISA 10.2 the LDS aperture is selected by addr[63:32] and the LDS offset
// is addr[31:0], so truncating a flat __shared__ pointer gives the wave-
// relative LDS byte offset the instruction's VDST expects.
// ---------------------------------------------------------------------------
__device__ __forceinline__ void async_cp16(void* lds_dst, const void* gsrc) {
  uint32_t dstoff = (uint32_t)(uintptr_t)lds_dst;
  uint64_t gaddr  = (uint64_t)(uintptr_t)gsrc;
  asm volatile("global_load_async_to_lds_b128 %0, %1, off"
               :: "v"(dstoff), "v"(gaddr) : "memory");
}
__device__ __forceinline__ void async_wait0() {
  asm volatile("s_wait_asynccnt 0x0" ::: "memory");
}

// ---------------------------------------------------------------------------
// posT[h][n][d] = rel_h[h,d,hh] + rel_w[h,d,w],  n = w*HH + hh
// ---------------------------------------------------------------------------
__global__ __launch_bounds__(256)
void pos_kernel(const float* __restrict__ rel_h, const float* __restrict__ rel_w,
                _Float16* __restrict__ posT) {
  int j  = blockIdx.x * 256 + threadIdx.x;   // over NHEAD*NN*DHD
  int d  = j & 63;
  int n  = (j >> 6) & (NN - 1);
  int h  = j >> 16;
  int hh = n & (HH - 1);
  int w  = n >> 5;
  int hd = h * DHD + d;
  posT[j] = (_Float16)(rel_h[hd * HH + hh] + rel_w[hd * WW + w]);
}

// ---------------------------------------------------------------------------
// Projection GEMM:  y[o,n] = sum_c Wm[o,c]*X[b,c,n] + bias[o]
// Block tile M=64 x N=128, K in 32-steps; 8 waves (2x4), 32x32 per wave.
// transposed=1 : Out is [b*NHEAD+h][n][d] f16 (head h = o>>6, d = o&63)
// transposed=0 : Out is [b][o][n] f16
// ---------------------------------------------------------------------------
__global__ __launch_bounds__(256)
void proj_gemm_kernel(const float* __restrict__ Wm, const float* __restrict__ bias,
                      const float* __restrict__ X, _Float16* __restrict__ Out,
                      int transposed) {
  __shared__ __align__(16) _Float16 As[64][40];    // [M][K] (+pad)
  __shared__ __align__(16) _Float16 Bs[128][40];   // [N][K] (+pad)

  const int b   = blockIdx.z;
  const int o0  = blockIdx.y * 64;
  const int n0  = blockIdx.x * 128;
  const int tid = threadIdx.x;
  const int wid = tid >> 5;
  const int wm  = wid >> 2;     // 0..1
  const int wn  = wid & 3;      // 0..3
  const float* Xb = X + (size_t)b * CC * NN;

  v8f zero = {};
  v8f acc[2][2];
#pragma unroll
  for (int i = 0; i < 2; ++i)
#pragma unroll
    for (int j = 0; j < 2; ++j) acc[i][j] = zero;

  const int ar = tid >> 2;          // A: row 0..63
  const int ak = (tid & 3) << 3;    // A: k-group base
  const int bn = tid & 127;         // B: column n 0..127

  for (int kk = 0; kk < CC; kk += 32) {
    __syncthreads();
    {  // A tile 64x32: one v8h per thread (two float4 global loads)
      const float* src = &Wm[(size_t)(o0 + ar) * CC + kk + ak];
      float4 w0 = *(const float4*)(src);
      float4 w1 = *(const float4*)(src + 4);
      v8h pk;
      pk[0] = (_Float16)w0.x; pk[1] = (_Float16)w0.y;
      pk[2] = (_Float16)w0.z; pk[3] = (_Float16)w0.w;
      pk[4] = (_Float16)w1.x; pk[5] = (_Float16)w1.y;
      pk[6] = (_Float16)w1.z; pk[7] = (_Float16)w1.w;
      *(v8h*)&As[ar][ak] = pk;
    }
#pragma unroll
    for (int i = 0; i < 2; ++i) {  // B tile as [n][k]: two v8h per thread
      int kg = (((tid >> 7) << 1) + i) << 3;   // 0,8,16,24
      v8h pk;
#pragma unroll
      for (int j = 0; j < 8; ++j)
        pk[j] = (_Float16)Xb[(size_t)(kk + kg + j) * NN + n0 + bn];
      *(v8h*)&Bs[bn][kg] = pk;
    }
    __syncthreads();

    v16h a0 = ld_frag(&As[wm * 32][0], 40);
    v16h a1 = ld_frag(&As[wm * 32 + 16][0], 40);
    v16h b0 = ld_frag(&Bs[wn * 32][0], 40);
    v16h b1 = ld_frag(&Bs[wn * 32 + 16][0], 40);
    acc[0][0] = __builtin_amdgcn_wmma_f32_16x16x32_f16(false, a0, false, b0, (short)0, acc[0][0], false, false);
    acc[0][1] = __builtin_amdgcn_wmma_f32_16x16x32_f16(false, a0, false, b1, (short)0, acc[0][1], false, false);
    acc[1][0] = __builtin_amdgcn_wmma_f32_16x16x32_f16(false, a1, false, b0, (short)0, acc[1][0], false, false);
    acc[1][1] = __builtin_amdgcn_wmma_f32_16x16x32_f16(false, a1, false, b1, (short)0, acc[1][1], false, false);
  }

  // D layout: lane<16 -> col=lane, v[r]=row r ; lane>=16 -> col=lane-16, row r+8
  const int lane = tid & 31;
  const int colL = lane & 15;
  const int rhi  = (lane >> 4) << 3;
  if (transposed) {
    const int hIdx = o0 >> 6;   // one head per 64-row tile
#pragma unroll
    for (int i = 0; i < 2; ++i)
#pragma unroll
      for (int j = 0; j < 2; ++j) {
        int dbase = wm * 32 + i * 16 + rhi;          // 8 consecutive d
        int ncol  = n0 + wn * 32 + j * 16 + colL;
        v8h pk;
#pragma unroll
        for (int r = 0; r < 8; ++r)
          pk[r] = (_Float16)(acc[i][j][r] + bias[o0 + dbase + r]);
        *(v8h*)&Out[((size_t)(b * NHEAD + hIdx) * NN + ncol) * DHD + dbase] = pk;
      }
  } else {
#pragma unroll
    for (int i = 0; i < 2; ++i)
#pragma unroll
      for (int j = 0; j < 2; ++j)
#pragma unroll
        for (int r = 0; r < 8; ++r) {
          int orow = o0 + wm * 32 + i * 16 + r + rhi;
          int ncol = n0 + wn * 32 + j * 16 + colL;
          Out[(size_t)(b * CC + orow) * NN + ncol] = (_Float16)(acc[i][j][r] + bias[orow]);
        }
  }
}

// ---------------------------------------------------------------------------
// Flash attention per (b, h, 64-query block); 4 waves, 16 queries per wave.
// logits[n,m] = [q;pos](:,n) . [k;q](:,m)   (K = 128, fused pos-bias GEMM)
// out[n,d]    = sum_m softmax(logits)[n,m] * v[d,m]
// Qt/Kt/posT are dims-contiguous [.][n][d]; Vf is [b][c][n] (dim-major).
// All tile staging uses GLOBAL_LOAD_ASYNC_TO_LDS_B128 (ASYNCcnt).
// ---------------------------------------------------------------------------
__global__ __launch_bounds__(128)
void flash_attn_kernel(const _Float16* __restrict__ Qt, const _Float16* __restrict__ Kt,
                       const _Float16* __restrict__ Vf, const _Float16* __restrict__ PosT,
                       float* __restrict__ Out) {
  __shared__ __align__(16) _Float16 Aq[64][136];    // [query][ q;pos dims ] (+pad)
  __shared__ __align__(16) _Float16 Bkq[64][136];   // [key]  [ k;q  dims ] (+pad)
  __shared__ __align__(16) _Float16 Vt[64][72];     // [dim][key] (+pad)
  __shared__ __align__(16) _Float16 Pl[4][16][72];  // per-wave P band [query][key]

  const int bid  = blockIdx.x;
  const int qb   = bid & 15;
  const int h    = (bid >> 4) & 7;
  const int b    = bid >> 7;
  const int n0   = qb * 64;
  const int tid  = threadIdx.x;
  const int wave = tid >> 5;
  const int lane = tid & 31;
  const int bh   = b * NHEAD + h;

  const _Float16* Qh = Qt + (size_t)bh * NN * DHD;
  const _Float16* Kh = Kt + (size_t)bh * NN * DHD;
  const _Float16* Vh = Vf + ((size_t)b * CC + h * DHD) * NN;
  const _Float16* Ph = PosT + (size_t)h * NN * DHD;

  {  // Aq: thread t copies 64 halves = 8 async b128, all contiguous
    const int nq = tid >> 1, part = tid & 1;
    const _Float16* src = part ? (Ph + (size_t)(n0 + nq) * DHD)
                               : (Qh + (size_t)(n0 + nq) * DHD);
    _Float16* dst = &Aq[nq][part * 64];
#pragma unroll
    for (int i = 0; i < 8; ++i)
      async_cp16(dst + i * 8, src + i * 8);
  }

  float rowmax[8], rowsum[8];
#pragma unroll
  for (int r = 0; r < 8; ++r) { rowmax[r] = -3.0e38f; rowsum[r] = 0.0f; }
  v8f zero = {};
  v8f oacc[4];
#pragma unroll
  for (int t = 0; t < 4; ++t) oacc[t] = zero;

  for (int m0 = 0; m0 < NN; m0 += 64) {
    __syncthreads();
    {  // Bkq row m = [ k(:,m) ; q(:,m) ]: 8 async b128 per thread
      const int m = tid >> 1, part = tid & 1;
      const _Float16* src = part ? (Qh + (size_t)(m0 + m) * DHD)
                                 : (Kh + (size_t)(m0 + m) * DHD);
      _Float16* dst = &Bkq[m][part * 64];
#pragma unroll
      for (int i = 0; i < 8; ++i)
        async_cp16(dst + i * 8, src + i * 8);
    }
    {  // Vt[d][m] straight from dim-major V: 4 async b128 per thread
      const int d = tid >> 1, half = tid & 1;
      const _Float16* src = Vh + (size_t)d * NN + m0 + half * 32;
      _Float16* dst = &Vt[d][half * 32];
#pragma unroll
      for (int i = 0; i < 4; ++i)
        async_cp16(dst + i * 8, src + i * 8);
    }
    if (m0 + 64 < NN)  // warm L2 for the next key block
      __builtin_prefetch(Kh + (size_t)(m0 + 64 + (tid >> 1)) * DHD, 0, 1);
    async_wait0();       // this wave's async LDS writes have landed
    __syncthreads();     // all waves' tiles visible

    // S band [16 x 64] = Aq(wave) x Bkq^T, K=128 in 4 steps
    v8f s[4];
#pragma unroll
    for (int j = 0; j < 4; ++j) s[j] = zero;
#pragma unroll
    for (int ks = 0; ks < 4; ++ks) {
      v16h a = ld_frag(&Aq[wave * 16][ks * 32], 136);
#pragma unroll
      for (int j = 0; j < 4; ++j) {
        v16h bf = ld_frag(&Bkq[j * 16][ks * 32], 136);
        s[j] = __builtin_amdgcn_wmma_f32_16x16x32_f16(false, a, false, bf, (short)0, s[j], false, false);
      }
    }

    // Online softmax; rows 0..7 live in lanes 0..15 (v[r]), rows 8..15 in 16..31,
    // so a 16-wide butterfly is exactly the per-row reduction over keys.
    float alpha[8];
#pragma unroll
    for (int r = 0; r < 8; ++r) {
      float mb = fmaxf(fmaxf(s[0][r], s[1][r]), fmaxf(s[2][r], s[3][r]));
#pragma unroll
      for (int off = 1; off < 16; off <<= 1)
        mb = fmaxf(mb, __shfl_xor(mb, off, 32));
      float mnew = fmaxf(rowmax[r], mb);
      alpha[r]  = __expf(rowmax[r] - mnew);
      rowmax[r] = mnew;
    }
    {
      const int colL = lane & 15;
      const int rhi  = (lane >> 4) << 3;
#pragma unroll
      for (int r = 0; r < 8; ++r) {
        float accp = 0.0f;
#pragma unroll
        for (int j = 0; j < 4; ++j) {
          float p = __expf(s[j][r] - rowmax[r]);
          Pl[wave][r + rhi][j * 16 + colL] = (_Float16)p;
          accp += p;
        }
#pragma unroll
        for (int off = 1; off < 16; off <<= 1)
          accp += __shfl_xor(accp, off, 32);
        rowsum[r] = rowsum[r] * alpha[r] + accp;
#pragma unroll
        for (int t = 0; t < 4; ++t) oacc[t][r] *= alpha[r];
      }
    }

    // O += P x V^T  (K = 64 keys in 2 steps); Pl is per-wave (same-wave LDS order)
#pragma unroll
    for (int ks = 0; ks < 2; ++ks) {
      v16h a = ld_frag(&Pl[wave][0][ks * 32], 72);
#pragma unroll
      for (int t = 0; t < 4; ++t) {
        v16h bf = ld_frag(&Vt[t * 16][ks * 32], 72);
        oacc[t] = __builtin_amdgcn_wmma_f32_16x16x32_f16(false, a, false, bf, (short)0, oacc[t], false, false);
      }
    }
  }

  const int colL = lane & 15;
  const int rhi  = (lane >> 4) << 3;
#pragma unroll
  for (int t = 0; t < 4; ++t)
#pragma unroll
    for (int r = 0; r < 8; ++r) {
      int d = t * 16 + colL;
      int n = n0 + wave * 16 + r + rhi;
      Out[((size_t)b * CC + h * DHD + d) * NN + n] = oacc[t][r] / rowsum[r];
    }
}

// ---------------------------------------------------------------------------
extern "C" void kernel_launch(void* const* d_in, const int* in_sizes, int n_in,
                              void* d_out, int out_size, void* d_ws, size_t ws_size,
                              hipStream_t stream) {
  const float* x     = (const float*)d_in[0];
  const float* wq    = (const float*)d_in[1];
  const float* bq    = (const float*)d_in[2];
  const float* wk    = (const float*)d_in[3];
  const float* bk    = (const float*)d_in[4];
  const float* wv    = (const float*)d_in[5];
  const float* bv    = (const float*)d_in[6];
  const float* rel_h = (const float*)d_in[7];
  const float* rel_w = (const float*)d_in[8];
  float* out = (float*)d_out;

  const size_t elems = (size_t)BB * CC * NN;     // == BB*NHEAD*NN*DHD
  _Float16* Qt  = (_Float16*)d_ws;               // [b*h][n][d]
  _Float16* Kt  = Qt + elems;                    // [b*h][n][d]
  _Float16* Vf  = Kt + elems;                    // [b][c][n]
  _Float16* Pos = Vf + elems;                    // [h][n][d]

  pos_kernel<<<(NHEAD * NN * DHD) / 256, 256, 0, stream>>>(rel_h, rel_w, Pos);

  dim3 g1(NN / 128, CC / 64, BB);
  proj_gemm_kernel<<<g1, 256, 0, stream>>>(wq, bq, x, Qt, 1);
  proj_gemm_kernel<<<g1, 256, 0, stream>>>(wk, bk, x, Kt, 1);
  proj_gemm_kernel<<<g1, 256, 0, stream>>>(wv, bv, x, Vf, 0);

  flash_attn_kernel<<<BB * NHEAD * (NN / 64), 128, 0, stream>>>(Qt, Kt, Vf, Pos, out);
}
